// Naive3DConvEncoder_84413287236161
// MI455X (gfx1250) — compile-verified, matrix operand order checked by reading, not run on hardware
//
#include <hip/hip_runtime.h>
#include <hip/hip_bf16.h>

typedef __attribute__((ext_vector_type(16))) _Float16 v16h;
typedef __attribute__((ext_vector_type(8)))  _Float16 v8h;
typedef __attribute__((ext_vector_type(8)))  float    v8f;

#define GRID_G 64
#define BATCH  32
#define NPTS   16384
#define FC_K   32768
#define KSPLIT 8
#define KCHUNK (FC_K / KSPLIT)

// ---------------- zero + scatter ----------------

__global__ void zero_f32(float* __restrict__ p, long n) {
  long i = (long)blockIdx.x * blockDim.x + threadIdx.x;
  if (i < n) p[i] = 0.0f;
}

__global__ void scatter_points(const float* __restrict__ coords, float* __restrict__ grid) {
  int i = blockIdx.x * blockDim.x + threadIdx.x;  // point index over B*N
  if (i >= BATCH * NPTS) return;
  int b = i / NPTS;
  const float* p = coords + (long)i * 3;
  float sx = (p[0] + 1.0f) * 0.5f * (GRID_G - 1);
  float sy = (p[1] + 1.0f) * 0.5f * (GRID_G - 1);
  float sz = (p[2] + 1.0f) * 0.5f * (GRID_G - 1);
  int cx = min(GRID_G - 1, max(0, (int)sx));   // trunc-toward-zero == astype(int32) for this range
  int cy = min(GRID_G - 1, max(0, (int)sy));
  int cz = min(GRID_G - 1, max(0, (int)sz));
  grid[(((long)b * GRID_G + cz) * GRID_G + cy) * GRID_G + cx] = 1.0f;
}

// ---------------- direct stride-2 3x3x3 conv + ReLU ----------------

template <int CI, int CO, int GI>
__global__ void conv3d_s2_relu(const float* __restrict__ x, const float* __restrict__ w,
                               const float* __restrict__ bias, float* __restrict__ y) {
  constexpr int GO = GI / 2;
  long tid = (long)blockIdx.x * blockDim.x + threadIdx.x;
  long total = (long)BATCH * CO * GO * GO * GO;
  if (tid >= total) return;
  int ow = (int)(tid % GO);
  int oh = (int)((tid / GO) % GO);
  int od = (int)((tid / ((long)GO * GO)) % GO);
  int oc = (int)((tid / ((long)GO * GO * GO)) % CO);
  int b  = (int)(tid / ((long)GO * GO * GO * CO));

  float acc = bias[oc];
  for (int ic = 0; ic < CI; ++ic) {
    const float* xb = x + ((long)b * CI + ic) * GI * GI * GI;
    const float* wb = w + ((long)oc * CI + ic) * 27;
#pragma unroll
    for (int kd = 0; kd < 3; ++kd) {
      int id = od * 2 + kd - 1;
      if (id < 0 || id >= GI) continue;
#pragma unroll
      for (int kh = 0; kh < 3; ++kh) {
        int ih = oh * 2 + kh - 1;
        if (ih < 0 || ih >= GI) continue;
#pragma unroll
        for (int kw = 0; kw < 3; ++kw) {
          int iw = ow * 2 + kw - 1;
          if (iw < 0 || iw >= GI) continue;
          acc += xb[((long)id * GI + ih) * GI + iw] * wb[(kd * 3 + kh) * 3 + kw];
        }
      }
    }
  }
  y[tid] = acc > 0.0f ? acc : 0.0f;
}

// ---------------- f32 -> f16 conversion (one pass, L2-resident afterwards) ----------------

__global__ void f32_to_f16(const float* __restrict__ src, _Float16* __restrict__ dst, long n) {
  long i = (long)blockIdx.x * blockDim.x + threadIdx.x;
  if (i < n) dst[i] = (_Float16)src[i];
}

// ---------------- FC as WMMA GEMM: out(32x128) = feat(32x32768) @ fc_w(128x32768)^T ----
// f16 operands pre-converted; one 16x16 tile per wave, split-K by KSPLIT, K unrolled by 2
// with independent accumulators; deterministic reduce kernel afterwards.

__device__ __forceinline__ v16h load_ab(const _Float16* __restrict__ row, int k0, int hi) {
  // 16-bit A/B 16x32 operand layout:
  //   VGPRs 0-3: K = k0 + hi*8 + {0..7};  VGPRs 4-7: K = k0 + 16 + hi*8 + {0..7}
  // Each run is 8 halfs = 16B, 16B-aligned -> global_load_b128.
  v8h lo = *(const v8h*)(row + k0 + hi * 8);
  v8h hv = *(const v8h*)(row + k0 + 16 + hi * 8);
  v16h out;
#pragma unroll
  for (int t = 0; t < 8; ++t) {
    out[t]     = lo[t];
    out[8 + t] = hv[t];
  }
  return out;
}

__global__ void fc_wmma_partial(const _Float16* __restrict__ feat, const _Float16* __restrict__ W,
                                float* __restrict__ partial) {
  const int lane = threadIdx.x;        // 0..31, one wave per block, EXEC all ones
  const int r    = lane & 15;
  const int hi   = lane >> 4;          // K half-select per ISA A/B layout
  const int tile = blockIdx.x >> 3;    // 0..15  (2 row-tiles x 8 col-tiles)
  const int kc   = blockIdx.x & 7;     // K chunk
  const int mt   = tile >> 3;          // 0..1
  const int nt   = tile & 7;           // 0..7

  const _Float16* arow = feat + (long)(mt * 16 + r) * FC_K;  // A row   (M = mt*16 + r)
  const _Float16* brow = W    + (long)(nt * 16 + r) * FC_K;  // W row -> B column (N = nt*16 + r)

  v8f c0 = {}, c1 = {};
  const int kend = (kc + 1) * KCHUNK;
  for (int k0 = kc * KCHUNK; k0 < kend; k0 += 64) {
    v16h a0 = load_ab(arow, k0, hi);
    v16h b0 = load_ab(brow, k0, hi);
    v16h a1 = load_ab(arow, k0 + 32, hi);
    v16h b1 = load_ab(brow, k0 + 32, hi);
    c0 = __builtin_amdgcn_wmma_f32_16x16x32_f16(false, a0, false, b0, (short)0, c0, false, false);
    c1 = __builtin_amdgcn_wmma_f32_16x16x32_f16(false, a1, false, b1, (short)0, c1, false, false);
  }
  v8f c = c0 + c1;

  // C/D layout: VGPR i -> M = i + hi*8 ; N = r
  float* pt = partial + (long)blockIdx.x * 256;
#pragma unroll
  for (int i = 0; i < 8; ++i) {
    int row_l = hi * 8 + i;
    pt[row_l * 16 + r] = c[i];
  }
}

__global__ void fc_reduce(const float* __restrict__ partial, const float* __restrict__ bias,
                          float* __restrict__ out) {
  int i = blockIdx.x * blockDim.x + threadIdx.x;  // 0..4095
  if (i >= BATCH * 128) return;
  int col = i & 127;
  int row = i >> 7;
  int mt = row >> 4, nt = col >> 4;
  int tile = mt * 8 + nt;
  int row_l = row & 15, col_l = col & 15;
  float s = bias[col];
#pragma unroll
  for (int kc = 0; kc < KSPLIT; ++kc)
    s += partial[(long)(tile * KSPLIT + kc) * 256 + row_l * 16 + col_l];
  out[i] = s;
}

// ---------------- launch ----------------

extern "C" void kernel_launch(void* const* d_in, const int* in_sizes, int n_in,
                              void* d_out, int out_size, void* d_ws, size_t ws_size,
                              hipStream_t stream) {
  const float* coords = (const float*)d_in[0];
  const float* w1  = (const float*)d_in[1];
  const float* b1  = (const float*)d_in[2];
  const float* w2  = (const float*)d_in[3];
  const float* b2  = (const float*)d_in[4];
  const float* w3  = (const float*)d_in[5];
  const float* b3  = (const float*)d_in[6];
  const float* fcw = (const float*)d_in[7];
  const float* fcb = (const float*)d_in[8];
  float* out = (float*)d_out;
  float* ws  = (float*)d_ws;

  const long NGRID = (long)BATCH * GRID_G * GRID_G * GRID_G;   // 8388608
  const long NY1   = (long)BATCH * 16 * 32 * 32 * 32;          // 16777216
  const long NY2   = (long)BATCH * 32 * 16 * 16 * 16;          // 4194304
  const long NY3   = (long)BATCH * 64 * 8 * 8 * 8;             // 1048576 (= 32 x 32768 feat)
  const long NFCW  = (long)128 * FC_K;                         // 4194304

  float* grid   = ws;
  float* y1     = grid + NGRID;
  float* y2     = y1 + NY1;
  float* y3     = y2 + NY2;
  float* fcpart = y3 + NY3;                                    // 128*256 = 32768 f32
  _Float16* feat_h = (_Float16*)(fcpart + 32768);              // 16B-aligned (offset mult of 16)
  _Float16* fcw_h  = feat_h + NY3;

  zero_f32<<<(int)((NGRID + 255) / 256), 256, 0, stream>>>(grid, NGRID);

  scatter_points<<<(BATCH * NPTS + 255) / 256, 256, 0, stream>>>(coords, grid);

  conv3d_s2_relu<1, 16, 64><<<(int)((NY1 + 255) / 256), 256, 0, stream>>>(grid, w1, b1, y1);
  conv3d_s2_relu<16, 32, 32><<<(int)((NY2 + 255) / 256), 256, 0, stream>>>(y1, w2, b2, y2);
  conv3d_s2_relu<32, 64, 16><<<(int)((NY3 + 255) / 256), 256, 0, stream>>>(y2, w3, b3, y3);

  f32_to_f16<<<(int)((NY3 + 255) / 256), 256, 0, stream>>>(y3, feat_h, NY3);
  f32_to_f16<<<(int)((NFCW + 255) / 256), 256, 0, stream>>>(fcw, fcw_h, NFCW);

  fc_wmma_partial<<<16 * KSPLIT, 32, 0, stream>>>(feat_h, fcw_h, fcpart);
  fc_reduce<<<(BATCH * 128 + 255) / 256, 256, 0, stream>>>(fcpart, fcb, out);
}
